// CausalSelfAttention_6631429505192
// MI455X (gfx1250) — matrix operand order repaired and checked by reading, not compiled, not run on hardware
//
#include <hip/hip_runtime.h>

typedef __bf16 bf16;
typedef __attribute__((ext_vector_type(16))) __bf16 v16bf;
typedef __attribute__((ext_vector_type(8)))  float  v8f;

union BF16x16 { v16bf v; uint4 u[2]; };

__device__ __forceinline__ v8f wmma_bf16(v16bf a, v16bf b, v8f c) {
  return __builtin_amdgcn_wmma_f32_16x16x32_bf16(false, a, false, b, (short)0, c, false, false);
}

// ---- gfx1250 async global->LDS path (feature-gated, sync fallback) ----
#if defined(__has_builtin)
#  if __has_builtin(__builtin_amdgcn_global_load_async_to_lds_b128) && \
      __has_builtin(__builtin_amdgcn_s_wait_asynccnt)
#    define HAVE_ASYNC_LDS 1
#  endif
#endif
#ifndef HAVE_ASYNC_LDS
#  define HAVE_ASYNC_LDS 0
#endif

#if HAVE_ASYNC_LDS
typedef int async_v4i __attribute__((vector_size(16)));
typedef __attribute__((address_space(1))) async_v4i* async_gp;   // global (__device__)
typedef __attribute__((address_space(3))) async_v4i* async_lp;   // LDS (__shared__)
#endif

__device__ __forceinline__ void copy16_to_lds(const bf16* g, bf16* l) {
#if HAVE_ASYNC_LDS
  __builtin_amdgcn_global_load_async_to_lds_b128((async_gp)(void*)g, (async_lp)(void*)l, 0, 0);
#else
  *(uint4*)l = *(const uint4*)g;
#endif
}
__device__ __forceinline__ void lds_copy_wait() {
#if HAVE_ASYNC_LDS
  __builtin_amdgcn_s_wait_asynccnt(0);
#endif
}

// problem constants
#define TSEQ 2048
#define CDIM 2048
#define KVDIM 512

// ---------------------------------------------------------------- conversion
__global__ __launch_bounds__(256) void cvt_f32_bf16(const float* __restrict__ s,
                                                    bf16* __restrict__ d, int n) {
  int i = blockIdx.x * 256 + threadIdx.x;
  if (i < n) d[i] = (bf16)s[i];
}

// ---------------------------------------------------------------- GEMM  Y = X * W^T
// X: [M,K] bf16 (lda), W: [N,K] bf16 (ldb).
// 256 thr = 8 waves; wave -> 64(M) x 32(N) tile (4x2 WMMA accums);
// block -> 128(M) x 128(N).
template <bool F32OUT>
__global__ __launch_bounds__(256) void gemm_bf16(const bf16* __restrict__ A, int lda,
                                                 const bf16* __restrict__ Bw, int ldb,
                                                 bf16* __restrict__ Cb,
                                                 float* __restrict__ Cf,
                                                 int ldc, int K) {
  const int lane   = threadIdx.x & 31;
  const int wave   = threadIdx.x >> 5;
  const int lane15 = lane & 15;
  const int hi     = lane >> 4;
  const int m0 = blockIdx.y * 128 + (wave & 1) * 64;
  const int n0 = blockIdx.x * 128 + (wave >> 1) * 32;

  v8f acc[4][2] = {};

  const bf16* arow[4];
#pragma unroll
  for (int i = 0; i < 4; ++i)
    arow[i] = A + (size_t)(m0 + i * 16 + lane15) * lda + hi * 8;
  const bf16* brow[2];
#pragma unroll
  for (int j = 0; j < 2; ++j)
    brow[j] = Bw + (size_t)(n0 + j * 16 + lane15) * ldb + hi * 8;

  for (int kb = 0; kb < K; kb += 32) {
    __builtin_prefetch(arow[0] + kb + 512, 0, 1);
    __builtin_prefetch(brow[0] + kb + 512, 0, 1);
    BF16x16 a[4], b[2];
#pragma unroll
    for (int i = 0; i < 4; ++i) {
      a[i].u[0] = *(const uint4*)(arow[i] + kb);
      a[i].u[1] = *(const uint4*)(arow[i] + kb + 16);
    }
#pragma unroll
    for (int j = 0; j < 2; ++j) {
      b[j].u[0] = *(const uint4*)(brow[j] + kb);
      b[j].u[1] = *(const uint4*)(brow[j] + kb + 16);
    }
#pragma unroll
    for (int i = 0; i < 4; ++i)
#pragma unroll
      for (int j = 0; j < 2; ++j)
        acc[i][j] = wmma_bf16(a[i].v, b[j].v, acc[i][j]);
  }

#pragma unroll
  for (int i = 0; i < 4; ++i) {
#pragma unroll
    for (int j = 0; j < 2; ++j) {
#pragma unroll
      for (int r = 0; r < 8; ++r) {
        int row = m0 + i * 16 + hi * 8 + r;
        int col = n0 + j * 16 + lane15;
        if (F32OUT) Cf[(size_t)row * ldc + col] = acc[i][j][r];
        else        Cb[(size_t)row * ldc + col] = (bf16)acc[i][j][r];
      }
    }
  }
}

// ---------------------------------------------------------------- RoPE (16 dims only)
#define LN_10000 9.210340371976184f

__device__ __forceinline__ void rope16(bf16* x, int t) {
#pragma unroll
  for (int i = 0; i < 8; ++i) {
    float f = __expf(-(float)i * 0.125f * LN_10000);
    float ang = (float)t * f;
    float c = __cosf(ang), s = __sinf(ang);
    float x1 = (float)x[i], x2 = (float)x[i + 8];
    x[i]     = (bf16)(x1 * c - x2 * s);
    x[i + 8] = (bf16)(x2 * c + x1 * s);
  }
}

__global__ __launch_bounds__(256) void rope_q(bf16* __restrict__ Q) {
  int idx = blockIdx.x * 256 + threadIdx.x;  // over M*16
  int h = idx & 15;
  int row = idx >> 4;  // b*T + t
  rope16(Q + (size_t)row * CDIM + h * 128, row & (TSEQ - 1));
}

__global__ __launch_bounds__(256) void rope_k(bf16* __restrict__ Kb) {
  int idx = blockIdx.x * 256 + threadIdx.x;  // over M*4
  int h = idx & 3;
  int row = idx >> 2;
  rope16(Kb + (size_t)row * KVDIM + h * 128, row & (TSEQ - 1));
}

// second output: repeat(v,4,axis=1)[:, :4] == kv-head 0 replicated 4x, [b,j,t,d] f32
__global__ __launch_bounds__(256) void v_out_kernel(const bf16* __restrict__ Vb,
                                                    float* __restrict__ out2) {
  int idx = blockIdx.x * 256 + threadIdx.x;  // over M*512
  int c = idx & 511;
  int row = idx >> 9;                 // b*T + t
  int b = row >> 11, t = row & (TSEQ - 1);
  int j = c >> 7, d = c & 127;
  out2[(((size_t)(b * 4 + j)) * TSEQ + t) * 128 + d] = (float)Vb[(size_t)row * KVDIM + d];
}

// ---------------------------------------------------------------- flash attention
// block = (b, h, 128-query tile); 8 waves, wave = 16 query rows.
__global__ __launch_bounds__(256) void flash_attn(const bf16* __restrict__ Q,
                                                  const bf16* __restrict__ Kb,
                                                  const bf16* __restrict__ Vb,
                                                  bf16* __restrict__ Obuf,
                                                  const float* __restrict__ gain) {
  __shared__ alignas(16) bf16 sK[32 * 128];        // [key][d]
  __shared__ alignas(16) bf16 sVt[128 * 32];       // [d][key]  (transposed)
  __shared__ alignas(16) bf16 sP[8][16 * 32];      // per-wave P bounce

  const int lane = threadIdx.x & 31, wave = threadIdx.x >> 5;
  const int lane15 = lane & 15, hi = lane >> 4;
  const int bid = blockIdx.x;
  const int qt = bid & 15;
  const int h  = (bid >> 4) & 15;
  const int b  = bid >> 8;
  const int kvh = h >> 2;
  const int tile0 = qt * 128;
  const int qbase = tile0 + wave * 16;
  const float scaleh = 0.08838834764831845f * gain[h];  // gain folded into score scale

  const bf16* Qbase = Q  + (size_t)b * TSEQ * CDIM  + (size_t)h * 128;
  const bf16* Kbase = Kb + (size_t)b * TSEQ * KVDIM + (size_t)kvh * 128;
  const bf16* Vbase = Vb + (size_t)b * TSEQ * KVDIM + (size_t)kvh * 128;

  // preload Q fragments (16 rows x 128 head dim = 4 A-fragments)
  v16bf aq[4];
  {
    const bf16* qr = Qbase + (size_t)(qbase + lane15) * CDIM;
#pragma unroll
    for (int kk = 0; kk < 4; ++kk) {
      BF16x16 t;
      t.u[0] = *(const uint4*)(qr + kk * 32 + hi * 8);
      t.u[1] = *(const uint4*)(qr + kk * 32 + 16 + hi * 8);
      aq[kk] = t.v;
    }
  }

  v8f o[4] = {};
  float mrun[8], lrun[8];
#pragma unroll
  for (int r = 0; r < 8; ++r) { mrun[r] = -1e30f; lrun[r] = 0.0f; }

  const int nkb = (tile0 + 128) / 32;

  for (int kb = 0; kb < nkb; ++kb) {
    const int kbase = kb * 32;
    // ---- cooperative staging: K row-major (async), V transposed (manual)
    {
      int kr = threadIdx.x >> 3;          // 0..31 key row
      int ko = (threadIdx.x & 7) * 16;    // 0..112 dim offset
      const bf16* ks = Kbase + (size_t)(kbase + kr) * KVDIM + ko;
      copy16_to_lds(ks,     sK + kr * 128 + ko);
      copy16_to_lds(ks + 8, sK + kr * 128 + ko + 8);
      const bf16* vs = Vbase + (size_t)(kbase + kr) * KVDIM + ko;
      alignas(16) bf16 tmp[16];
      *(uint4*)(tmp)     = *(const uint4*)(vs);
      *(uint4*)(tmp + 8) = *(const uint4*)(vs + 8);
#pragma unroll
      for (int i = 0; i < 16; ++i) sVt[(ko + i) * 32 + kr] = tmp[i];
    }
    lds_copy_wait();
    __syncthreads();

    if (kbase <= qbase + 15) {  // wave-uniform causal skip
      // ---- scores: S(16x32) = Q(16x128) * K^T
      v8f s0{}, s1{};
#pragma unroll
      for (int kk = 0; kk < 4; ++kk) {
        BF16x16 bk0, bk1;
        const bf16* kp0 = sK + (size_t)lane15 * 128 + kk * 32 + hi * 8;
        const bf16* kp1 = sK + (size_t)(16 + lane15) * 128 + kk * 32 + hi * 8;
        bk0.u[0] = *(const uint4*)(kp0);  bk0.u[1] = *(const uint4*)(kp0 + 16);
        bk1.u[0] = *(const uint4*)(kp1);  bk1.u[1] = *(const uint4*)(kp1 + 16);
        s0 = wmma_bf16(aq[kk], bk0.v, s0);
        s1 = wmma_bf16(aq[kk], bk1.v, s1);
      }
      // ---- online softmax (rows live in 16-lane halves)
#pragma unroll
      for (int r = 0; r < 8; ++r) {
        int qrow = qbase + r + 8 * hi;
        int key0 = kbase + lane15;
        int key1 = key0 + 16;
        float v0 = (key0 <= qrow) ? s0[r] * scaleh : -1e30f;
        float v1 = (key1 <= qrow) ? s1[r] * scaleh : -1e30f;
        float mx = fmaxf(v0, v1);
        mx = fmaxf(mx, __shfl_xor(mx, 1, 32));
        mx = fmaxf(mx, __shfl_xor(mx, 2, 32));
        mx = fmaxf(mx, __shfl_xor(mx, 4, 32));
        mx = fmaxf(mx, __shfl_xor(mx, 8, 32));
        float mnew = fmaxf(mrun[r], mx);
        float a  = __expf(mrun[r] - mnew);
        float p0 = __expf(v0 - mnew);
        float p1 = __expf(v1 - mnew);
        float rs = p0 + p1;
        rs += __shfl_xor(rs, 1, 32);
        rs += __shfl_xor(rs, 2, 32);
        rs += __shfl_xor(rs, 4, 32);
        rs += __shfl_xor(rs, 8, 32);
        mrun[r] = mnew;
        lrun[r] = lrun[r] * a + rs;
        int m = r + 8 * hi;
        sP[wave][m * 32 + lane15]      = (bf16)p0;
        sP[wave][m * 32 + 16 + lane15] = (bf16)p1;
#pragma unroll
        for (int jd = 0; jd < 4; ++jd) o[jd][r] *= a;
      }
      // ---- re-load P in A-layout and do O += P * V
      BF16x16 ap;
      const bf16* pp = &sP[wave][lane15 * 32];
      ap.u[0] = *(const uint4*)(pp + hi * 8);
      ap.u[1] = *(const uint4*)(pp + 16 + hi * 8);
#pragma unroll
      for (int jd = 0; jd < 4; ++jd) {
        BF16x16 bv;
        const bf16* vp = sVt + (size_t)(jd * 16 + lane15) * 32 + hi * 8;
        bv.u[0] = *(const uint4*)(vp);  bv.u[1] = *(const uint4*)(vp + 16);
        o[jd] = wmma_bf16(ap.v, bv.v, o[jd]);
      }
    }
    __syncthreads();
  }

  // ---- epilogue: normalize and store bf16 [b,t,h,d]
#pragma unroll
  for (int jd = 0; jd < 4; ++jd) {
#pragma unroll
    for (int r = 0; r < 8; ++r) {
      int qrow = qbase + r + 8 * hi;
      float val = o[jd][r] / lrun[r];
      Obuf[(size_t)(b * TSEQ + qrow) * CDIM + h * 128 + jd * 16 + lane15] = (bf16)val;
    }
  }
}

// ---------------------------------------------------------------- launch
extern "C" void kernel_launch(void* const* d_in, const int* in_sizes, int n_in,
                              void* d_out, int out_size, void* d_ws, size_t ws_size,
                              hipStream_t stream) {
  (void)in_sizes; (void)n_in; (void)out_size; (void)ws_size;
  const float* x    = (const float*)d_in[0];
  const float* qw   = (const float*)d_in[1];
  const float* kw   = (const float*)d_in[2];
  const float* vw   = (const float*)d_in[3];
  const float* ow   = (const float*)d_in[4];
  const float* gain = (const float*)d_in[5];
  float* out   = (float*)d_out;
  float* out_v = out + (size_t)4 * TSEQ * CDIM;

  const int Bn = 4, T = TSEQ, C = CDIM, KV = KVDIM;
  const int M = Bn * T;  // 8192

  char* ws = (char*)d_ws;
  size_t off = 0;
  auto alloc = [&](size_t bytes) { void* p = ws + off; off += (bytes + 255) & ~(size_t)255; return p; };
  bf16* Xb = (bf16*)alloc((size_t)M * C * 2);
  bf16* Wq = (bf16*)alloc((size_t)C * C * 2);
  bf16* Wk = (bf16*)alloc((size_t)KV * C * 2);
  bf16* Wv = (bf16*)alloc((size_t)KV * C * 2);
  bf16* Wo = (bf16*)alloc((size_t)C * C * 2);
  bf16* Qb = (bf16*)alloc((size_t)M * C * 2);
  bf16* Kb = (bf16*)alloc((size_t)M * KV * 2);
  bf16* Vb = (bf16*)alloc((size_t)M * KV * 2);
  bf16* Ab = (bf16*)alloc((size_t)M * C * 2);

  cvt_f32_bf16<<<(M * C + 255) / 256, 256, 0, stream>>>(x, Xb, M * C);
  cvt_f32_bf16<<<(C * C + 255) / 256, 256, 0, stream>>>(qw, Wq, C * C);
  cvt_f32_bf16<<<(KV * C + 255) / 256, 256, 0, stream>>>(kw, Wk, KV * C);
  cvt_f32_bf16<<<(KV * C + 255) / 256, 256, 0, stream>>>(vw, Wv, KV * C);
  cvt_f32_bf16<<<(C * C + 255) / 256, 256, 0, stream>>>(ow, Wo, C * C);

  gemm_bf16<false><<<dim3(C / 128, M / 128), 256, 0, stream>>>(Xb, C, Wq, C, Qb, nullptr, C, C);
  gemm_bf16<false><<<dim3(KV / 128, M / 128), 256, 0, stream>>>(Xb, C, Wk, C, Kb, nullptr, KV, C);
  gemm_bf16<false><<<dim3(KV / 128, M / 128), 256, 0, stream>>>(Xb, C, Wv, C, Vb, nullptr, KV, C);

  rope_q<<<M * 16 / 256, 256, 0, stream>>>(Qb);
  rope_k<<<M * 4 / 256, 256, 0, stream>>>(Kb);
  v_out_kernel<<<M * 512 / 256, 256, 0, stream>>>(Vb, out_v);

  flash_attn<<<Bn * 16 * (T / 128), 256, 0, stream>>>(Qb, Kb, Vb, Ab, gain);

  gemm_bf16<true><<<dim3(C / 128, M / 128), 256, 0, stream>>>(Ab, C, Wo, C, nullptr, out, C, C);
}
